// PairwiseBatchedDistance_58119497449848
// MI455X (gfx1250) — matrix operand order repaired and checked by reading, not compiled, not run on hardware
//
#include <hip/hip_runtime.h>

// ---------------------------------------------------------------------------
// Pairwise squared L2 distance:  d2 = ||x||^2 + ||y||^2 - 2 * x @ y^T
// x: [B,D] f32, y: [N,D] f32, out: [B,N] f32.   B=4096, N=16384, D=1024.
//
// Pipeline:
//   1) sqnorm_kernel:  ||x||^2, ||y||^2  (wave-per-row reduction)      -> ws
//   2) convert_split_kernel: f32 -> (bf16 hi, bf16 lo) K-panels        -> ws
//      layout: [kt][row][hi 64B | lo 64B]  (contiguous 128B per row/chunk)
//   3) pdist2_wmma_kernel: bf16x3 split GEMM on v_wmma_f32_16x16x32_bf16,
//      async global->LDS staging (double buffered, ASYNCcnt), 32x64 macro-tile
//      per wave (24 ds_load_b128 : 24 WMMA), fused -2xy+xs+ys NT epilogue.
// ---------------------------------------------------------------------------

typedef __attribute__((ext_vector_type(16))) __bf16 v16bf;
typedef __attribute__((ext_vector_type(8)))  __bf16 v8bf;
typedef __attribute__((ext_vector_type(8)))  float  v8f;
typedef int v4i __attribute__((vector_size(16)));
typedef __attribute__((address_space(1))) v4i* gv4i_p;   // global int4*
typedef __attribute__((address_space(3))) v4i* lv4i_p;   // LDS int4*

#define D_DIM 1024
#define KT_N  (D_DIM / 32)   // 32 K-chunks
#define TILE  128            // output tile per block (128x128)
#define ROWB  128            // bytes per row per K-chunk in global panel (64 hi + 64 lo)
#define LPITCH 144           // LDS row pitch in bytes (16B-aligned, 36 banks -> no conflicts)

#if __has_builtin(__builtin_amdgcn_global_load_async_to_lds_b128)
#define HAVE_ASYNC 1
#else
#define HAVE_ASYNC 0
#endif

#if HAVE_ASYNC
#if __has_builtin(__builtin_amdgcn_s_wait_asynccnt)
#define ASYNC_WAIT(N) __builtin_amdgcn_s_wait_asynccnt(N)
#else
#define ASYNC_WAIT(N) asm volatile("s_wait_asynccnt " #N)
#endif
#else
#define ASYNC_WAIT(N)
#endif

static __device__ __forceinline__ void async16(const unsigned char* g, unsigned char* l) {
#if HAVE_ASYNC
  __builtin_amdgcn_global_load_async_to_lds_b128((gv4i_p)g, (lv4i_p)l, 0, 0);
#else
  *(uint4*)l = *(const uint4*)g;   // sync fallback
#endif
}

static __device__ __forceinline__ unsigned short f32_bf16_rne(float f) {
  unsigned u = __builtin_bit_cast(unsigned, f);
  u += 0x7FFFu + ((u >> 16) & 1u);              // round-to-nearest-even
  return (unsigned short)(u >> 16);
}
static __device__ __forceinline__ float bf16_f32(unsigned short h) {
  unsigned u = ((unsigned)h) << 16;
  return __builtin_bit_cast(float, u);
}

struct HL { uint2 hi, lo; };
static __device__ __forceinline__ HL split4(float4 q) {
  unsigned short h0 = f32_bf16_rne(q.x), h1 = f32_bf16_rne(q.y);
  unsigned short h2 = f32_bf16_rne(q.z), h3 = f32_bf16_rne(q.w);
  unsigned short l0 = f32_bf16_rne(q.x - bf16_f32(h0));
  unsigned short l1 = f32_bf16_rne(q.y - bf16_f32(h1));
  unsigned short l2 = f32_bf16_rne(q.z - bf16_f32(h2));
  unsigned short l3 = f32_bf16_rne(q.w - bf16_f32(h3));
  HL r;
  r.hi = make_uint2((unsigned)h0 | ((unsigned)h1 << 16),
                    (unsigned)h2 | ((unsigned)h3 << 16));
  r.lo = make_uint2((unsigned)l0 | ((unsigned)l1 << 16),
                    (unsigned)l2 | ((unsigned)l3 << 16));
  return r;
}

// ---- row squared-norm: one wave32 per row, D = 1024 -----------------------
__global__ __launch_bounds__(256)
void sqnorm_kernel(const float* __restrict__ V, float* __restrict__ OUT, int rows) {
  const int wave = threadIdx.x >> 5;
  const int lane = threadIdx.x & 31;
  const int row  = blockIdx.x * 8 + wave;
  if (row >= rows) return;
  const float* p = V + (size_t)row * D_DIM;
  float s = 0.f;
  #pragma unroll
  for (int i = 0; i < 8; ++i) {
    float4 q = *(const float4*)(p + (size_t)(lane + i * 32) * 4);
    s += q.x * q.x + q.y * q.y + q.z * q.z + q.w * q.w;
  }
  #pragma unroll
  for (int off = 16; off > 0; off >>= 1) s += __shfl_xor(s, off, 32);
  if (lane == 0) OUT[row] = s;
}

// ---- one-time f32 -> bf16 hi/lo K-panel conversion ------------------------
// Panel layout: byte offset ((kt*rows + row) * 128) + [hi: kin*2 | lo: 64+kin*2]
__global__ __launch_bounds__(256)
void convert_split_kernel(const float* __restrict__ V, unsigned char* __restrict__ P,
                          int rows) {
  const size_t idx = (size_t)blockIdx.x * 256 + threadIdx.x;  // float4 index
  const int row = (int)(idx >> 8);        // 256 float4 per row
  const int q   = (int)(idx & 255);
  if (row >= rows) return;
  const int k   = q * 4;
  const int kt  = k >> 5;
  const int kin = k & 31;
  float4 f = *(const float4*)(V + (size_t)row * D_DIM + k);
  HL s = split4(f);
  unsigned char* base = P + ((size_t)kt * rows + row) * ROWB;
  *(uint2*)(base + kin * 2)      = s.hi;
  *(uint2*)(base + 64 + kin * 2) = s.lo;
}

// ---- stage one 128-row K-chunk (A and B) into LDS -------------------------
// 8 x async b128 per thread (4 A + 4 B) => ASYNCcnt 8 per wave per chunk.
static __device__ __forceinline__ void stage_chunk(const unsigned char* __restrict__ ga,
                                                   const unsigned char* __restrict__ gb,
                                                   unsigned char* la, unsigned char* lb,
                                                   int t) {
  #pragma unroll
  for (int j = 0; j < 4; ++j) {
    const int ca = t + j * 256;          // 0..1023 chunk id
    const int r  = ca >> 3;              // row 0..127
    const int ib = (ca & 7) * 16;        // byte-in-row 0..112
    async16(ga + (size_t)r * ROWB + ib, la + r * LPITCH + ib);
  }
  #pragma unroll
  for (int j = 0; j < 4; ++j) {
    const int ca = t + j * 256;
    const int r  = ca >> 3;
    const int ib = (ca & 7) * 16;
    async16(gb + (size_t)r * ROWB + ib, lb + r * LPITCH + ib);
  }
}

static __device__ __forceinline__ v16bf cat8(v8bf a, v8bf b) {
  return __builtin_shufflevector(a, b, 0,1,2,3,4,5,6,7,8,9,10,11,12,13,14,15);
}

// ---- main fused distance kernel -------------------------------------------
// 8 waves tile the 128x128 block as 4x2 macro-tiles of 32(rows) x 64(cols).
// Per wave per K-chunk: 8 A-frag + 16 B-frag ds_load_b128 feed 24 WMMAs.
__global__ __launch_bounds__(256)
void pdist2_wmma_kernel(const unsigned char* __restrict__ XP,
                        const unsigned char* __restrict__ YP,
                        const float* __restrict__ XS, const float* __restrict__ YS,
                        float* __restrict__ OUT, int Brows, int Ncols) {
  __shared__ __attribute__((aligned(16))) unsigned char sA[2][TILE * LPITCH];
  __shared__ __attribute__((aligned(16))) unsigned char sB[2][TILE * LPITCH];

  const int t    = threadIdx.x;
  const int wave = t >> 5;
  const int lane = t & 31;
  const int l16  = lane & 15;
  const int hiH  = lane >> 4;

  const int mr = wave >> 1;       // macro-row 0..3  -> rows  mr*32 + {0,16}
  const int mc = wave & 1;        // macro-col 0..1  -> cols  mc*64 + {0,16,32,48}

  const int rowBase = blockIdx.y * TILE;   // x rows
  const int colBase = blockIdx.x * TILE;   // y rows (= output cols)

  const unsigned char* Abase = XP + (size_t)rowBase * ROWB;
  const unsigned char* Bbase = YP + (size_t)colBase * ROWB;
  const size_t Astep = (size_t)Brows * ROWB;   // panel stride per K-chunk
  const size_t Bstep = (size_t)Ncols * ROWB;

  v8f acc[2][4];
  {
    v8f z = {0.f, 0.f, 0.f, 0.f, 0.f, 0.f, 0.f, 0.f};
    #pragma unroll
    for (int r = 0; r < 2; ++r)
      #pragma unroll
      for (int n = 0; n < 4; ++n) acc[r][n] = z;
  }

  // Fragment byte offsets per ISA layouts:
  const int arow0 = (mr * 32 + l16) * LPITCH;       // row-tile 0
  const int arow1 = arow0 + 16 * LPITCH;            // row-tile 1
  const int aoff  = hiH ? 16 : 0;   // A: lanes 0-15 K{0-7,16-23}, 16-31 K{8-15,24-31}
  const int boff  = hiH ? 16 : 0;   // B: lanes 0-15 K{0-7,16-23}, 16-31 K{8-15,24-31}

  stage_chunk(Abase, Bbase, sA[0], sB[0], t);   // prologue: chunk 0

  #pragma unroll 1
  for (int kt = 0; kt < KT_N; ++kt) {
    const int cur = kt & 1;
    if (kt + 1 < KT_N) {
      stage_chunk(Abase + (size_t)(kt + 1) * Astep,
                  Bbase + (size_t)(kt + 1) * Bstep,
                  sA[cur ^ 1], sB[cur ^ 1], t);
      ASYNC_WAIT(8);                  // current chunk landed (next still inflight)
    } else {
      ASYNC_WAIT(0);
    }
    __syncthreads();                  // all waves' data visible

    const unsigned char* A  = sA[cur];
    const unsigned char* Bp = sB[cur];

    v16bf aHi0 = cat8(*(const v8bf*)(A + arow0 + aoff),
                      *(const v8bf*)(A + arow0 + aoff + 32));
    v16bf aLo0 = cat8(*(const v8bf*)(A + arow0 + 64 + aoff),
                      *(const v8bf*)(A + arow0 + 64 + aoff + 32));
    v16bf aHi1 = cat8(*(const v8bf*)(A + arow1 + aoff),
                      *(const v8bf*)(A + arow1 + aoff + 32));
    v16bf aLo1 = cat8(*(const v8bf*)(A + arow1 + 64 + aoff),
                      *(const v8bf*)(A + arow1 + 64 + aoff + 32));

    #pragma unroll
    for (int n = 0; n < 4; ++n) {
      const unsigned char* Bq = Bp + (mc * 64 + n * 16 + l16) * LPITCH;
      v16bf bHi = cat8(*(const v8bf*)(Bq + boff),
                       *(const v8bf*)(Bq + boff + 32));
      v16bf bLo = cat8(*(const v8bf*)(Bq + 64 + boff),
                       *(const v8bf*)(Bq + 64 + boff + 32));
      acc[0][n] = __builtin_amdgcn_wmma_f32_16x16x32_bf16(false, aHi0, false, bHi,
                                                          (short)0, acc[0][n], false, false);
      acc[0][n] = __builtin_amdgcn_wmma_f32_16x16x32_bf16(false, aHi0, false, bLo,
                                                          (short)0, acc[0][n], false, false);
      acc[0][n] = __builtin_amdgcn_wmma_f32_16x16x32_bf16(false, aLo0, false, bHi,
                                                          (short)0, acc[0][n], false, false);
      acc[1][n] = __builtin_amdgcn_wmma_f32_16x16x32_bf16(false, aHi1, false, bHi,
                                                          (short)0, acc[1][n], false, false);
      acc[1][n] = __builtin_amdgcn_wmma_f32_16x16x32_bf16(false, aHi1, false, bLo,
                                                          (short)0, acc[1][n], false, false);
      acc[1][n] = __builtin_amdgcn_wmma_f32_16x16x32_bf16(false, aLo1, false, bHi,
                                                          (short)0, acc[1][n], false, false);
    }
    __syncthreads();                  // compute done before buffer reuse
  }

  // Epilogue: d2 = xs + ys - 2*dot, nontemporal stores (don't pollute L2).
  // C layout: VGPR j -> M=j (lanes 0-15), M=j+8 (lanes 16-31); N = lane&15.
  #pragma unroll
  for (int r = 0; r < 2; ++r) {
    const int gr0 = rowBase + mr * 32 + r * 16 + (hiH ? 8 : 0);
    #pragma unroll
    for (int n = 0; n < 4; ++n) {
      const int gc = colBase + mc * 64 + n * 16 + l16;
      const float ysv = YS[gc];
      #pragma unroll
      for (int j = 0; j < 8; ++j) {
        const int gr = gr0 + j;
        __builtin_nontemporal_store(XS[gr] + ysv - 2.0f * acc[r][n][j],
                                    &OUT[(size_t)gr * Ncols + gc]);
      }
    }
  }
}

extern "C" void kernel_launch(void* const* d_in, const int* in_sizes, int n_in,
                              void* d_out, int out_size, void* d_ws, size_t ws_size,
                              hipStream_t stream) {
  const float* X = (const float*)d_in[0];   // [B, D]
  const float* Y = (const float*)d_in[1];   // [N, D]
  float* OUT = (float*)d_out;               // [B, N]

  const int Brows = in_sizes[0] / D_DIM;    // 4096
  const int Ncols = in_sizes[1] / D_DIM;    // 16384

  // Workspace layout: XS[B] f32 | YS[N] f32 | X panels | Y panels
  float* XS = (float*)d_ws;
  float* YS = XS + Brows;
  unsigned char* XP = (unsigned char*)(YS + Ncols);           // B * 4096 bytes
  unsigned char* YP = XP + (size_t)Brows * (size_t)D_DIM * 4; // N * 4096 bytes

  sqnorm_kernel<<<(Brows + 7) / 8, 256, 0, stream>>>(X, XS, Brows);
  sqnorm_kernel<<<(Ncols + 7) / 8, 256, 0, stream>>>(Y, YS, Ncols);
  convert_split_kernel<<<Brows, 256, 0, stream>>>(X, XP, Brows);
  convert_split_kernel<<<Ncols, 256, 0, stream>>>(Y, YP, Ncols);

  dim3 grid(Ncols / TILE, Brows / TILE);    // 128 x 32 = 4096 blocks
  pdist2_wmma_kernel<<<grid, 256, 0, stream>>>(XP, YP, XS, YS, OUT, Brows, Ncols);
}
